// AttentionBlock_76106820485812
// MI455X (gfx1250) — compile-verified, matrix operand order checked by reading, not run on hardware
//
#include <hip/hip_runtime.h>
#include <hip/hip_bf16.h>

// ---------------- problem constants ----------------
constexpr int NB = 8;       // batch
constexpr int NC = 512;     // channels
constexpr int NS = 1024;    // sequence = 32*32
constexpr int NHEADS = 8;
constexpr int NCH = 64;     // channels per head
constexpr int NBH = NB * NHEADS;  // 64
constexpr int NGROUPS = 32;
constexpr int NCPG = NC / NGROUPS; // 16
constexpr float GN_EPS = 1e-5f;

typedef __attribute__((ext_vector_type(16))) _Float16 v16h;
typedef __attribute__((ext_vector_type(8)))  _Float16 v8h;
typedef __attribute__((ext_vector_type(8)))  float    v8f;

// ---------------- WMMA helpers (gfx1250, wave32) ----------------
__device__ __forceinline__ v8f wmma16(v16h a, v16h b, v8f c) {
  return __builtin_amdgcn_wmma_f32_16x16x32_f16(
      /*neg_a=*/false, a, /*neg_b=*/false, b,
      /*c_mod=*/(short)0, c, /*reuse_a=*/false, /*reuse_b=*/false);
}

// Load one 16x32 f16 WMMA fragment whose K dimension is CONTIGUOUS in memory.
// Works for both operands (ISA 7.12.2 gives A and B the same per-lane K map):
//   lanes 0-15  : row (lane&15), K {0..7, 16..23}
//   lanes 16-31 : row (lane&15), K {8..15, 24..31}
// base points at (row 0, k 0); ld = element stride between rows.
// All call sites keep base/ld 16-byte aligned -> two global_load_b128 per call.
__device__ __forceinline__ v16h load_frag(const _Float16* base, long ld, int lane) {
  const _Float16* p = base + (long)(lane & 15) * ld + (((lane >> 4) & 1) << 3);
  v8h lo = *(const v8h*)p;
  v8h hi = *(const v8h*)(p + 16);
  return __builtin_shufflevector(lo, hi, 0, 1, 2, 3, 4, 5, 6, 7,
                                 8, 9, 10, 11, 12, 13, 14, 15);
}

// ---------------- K1: GroupNorm -> hT[s][c] (f16, c contiguous) ----------
__global__ void attn_gn_kernel(const float* __restrict__ x,
                               const float* __restrict__ gsc,
                               const float* __restrict__ gbi,
                               _Float16* __restrict__ hT) {
  int bg = blockIdx.x;
  int b = bg / NGROUPS, g = bg % NGROUPS;
  const float* xp = x + ((long)b * NC + (long)g * NCPG) * NS;
  float s = 0.f, s2 = 0.f;
  for (int i = threadIdx.x; i < NCPG * NS; i += blockDim.x) {
    float v = xp[i];
    s += v; s2 += v * v;
  }
  __shared__ float red0[8], red1[8];
  int lane = threadIdx.x & 31, wv = threadIdx.x >> 5;
#pragma unroll
  for (int m = 16; m >= 1; m >>= 1) {
    s  += __shfl_xor(s,  m, 32);
    s2 += __shfl_xor(s2, m, 32);
  }
  if (lane == 0) { red0[wv] = s; red1[wv] = s2; }
  __syncthreads();
  if (wv == 0) {
    float a = (lane < 8) ? red0[lane] : 0.f;
    float c = (lane < 8) ? red1[lane] : 0.f;
#pragma unroll
    for (int m = 4; m >= 1; m >>= 1) {
      a += __shfl_xor(a, m, 32);
      c += __shfl_xor(c, m, 32);
    }
    if (lane == 0) { red0[0] = a; red1[0] = c; }
  }
  __syncthreads();
  const float rn = 1.f / (float)(NCPG * NS);
  float mean = red0[0] * rn;
  float var  = red1[0] * rn - mean * mean;
  float rstd = rsqrtf(var + GN_EPS);
  float scl[NCPG], bia[NCPG];
#pragma unroll
  for (int cc = 0; cc < NCPG; ++cc) {
    float sc = gsc[g * NCPG + cc] * rstd;
    scl[cc] = sc;
    bia[cc] = gbi[g * NCPG + cc] - mean * sc;
  }
  _Float16* hp = hT + (long)b * NS * NC + (long)g * NCPG;
  for (int sidx = threadIdx.x; sidx < NS; sidx += blockDim.x) {
    v8h o0, o1;
#pragma unroll
    for (int cc = 0; cc < 8; ++cc)
      o0[cc] = (_Float16)(xp[(long)cc * NS + sidx] * scl[cc] + bia[cc]);
#pragma unroll
    for (int cc = 8; cc < 16; ++cc)
      o1[cc - 8] = (_Float16)(xp[(long)cc * NS + sidx] * scl[cc] + bia[cc]);
    *(v8h*)(hp + (long)sidx * NC)     = o0;
    *(v8h*)(hp + (long)sidx * NC + 8) = o1;
  }
}

// ---------------- K2: fp32 -> f16 convert (weights) ----------------
__global__ void attn_cvt_f16_kernel(const float* __restrict__ src,
                                    _Float16* __restrict__ dst, int n) {
  int i = blockIdx.x * blockDim.x + threadIdx.x;
  if (i < n) dst[i] = (_Float16)src[i];
}

// ---------------- K3: QKV GEMM, scatter into qT/kT/vbuf ------------------
// grid (8, 96, NB), block 128 (4 waves). Wave tile: 16(o) x 32(s), K=NC;
// the weight (A) fragment is shared across both 16-wide column tiles.
__global__ void attn_qkv_gemm_kernel(const _Float16* __restrict__ w16,
                                     const _Float16* __restrict__ hT,
                                     const float* __restrict__ bqkv,
                                     _Float16* __restrict__ qT,
                                     _Float16* __restrict__ kT,
                                     _Float16* __restrict__ vbuf) {
  int lane = threadIdx.x & 31, wv = threadIdx.x >> 5;
  int sb = (blockIdx.x * 4 + wv) * 32;
  int ob = blockIdx.y * 16;
  int b  = blockIdx.z;
  const _Float16* hB = hT + (long)b * NS * NC;
  v8f acc0 = {}, acc1 = {};
  for (int kb = 0; kb < NC; kb += 32) {
    __builtin_prefetch(hB + (long)sb * NC + kb + 32, 0, 1);
    v16h a  = load_frag(w16 + (long)ob * NC + kb, NC, lane);        // A: w[o][c]
    v16h b0 = load_frag(hB + (long)sb * NC + kb, NC, lane);         // B: hT[s][c]
    v16h b1 = load_frag(hB + (long)(sb + 16) * NC + kb, NC, lane);
    acc0 = wmma16(a, b0, acc0);
    acc1 = wmma16(a, b1, acc1);
  }
  const float kscale = 0.125f;  // 1/sqrt(64), folded into K
  int n = lane & 15, hb8 = ((lane >> 4) & 1) * 8;
  int o0 = ob + hb8;                 // 8 consecutive output rows per lane
  int head = o0 / (3 * NCH);         // 16-row tile never straddles a head
  int r0 = o0 % (3 * NCH);
  int bh = b * NHEADS + head;
  float bv[8];
#pragma unroll
  for (int e = 0; e < 8; ++e) bv[e] = bqkv[o0 + e];
#pragma unroll
  for (int half = 0; half < 2; ++half) {
    v8f acc = half ? acc1 : acc0;
    long srow = (long)bh * NS + sb + half * 16 + n;
    if (r0 < 2 * NCH) {              // q or k: contiguous v8h store
      v8h pk;
      float mul = (r0 < NCH) ? 1.f : kscale;
#pragma unroll
      for (int e = 0; e < 8; ++e)
        pk[e] = (_Float16)((acc[e] + bv[e]) * mul);
      _Float16* dst = (r0 < NCH) ? (qT + srow * NCH + r0)
                                 : (kT + srow * NCH + (r0 - NCH));
      *(v8h*)dst = pk;
    } else {                          // v: [bh][cc][t] scatter (8 x 2B)
      int cc0 = r0 - 2 * NCH;
#pragma unroll
      for (int e = 0; e < 8; ++e)
        vbuf[((long)bh * NCH + cc0 + e) * NS + sb + half * 16 + n] =
            (_Float16)(acc[e] + bv[e]);
    }
  }
}

// ---------------- K4: softmax stats over s, per t row --------------------
// T[t,s] = sum_c kT[t][c]*qT[s][c]; stats (max, 1/sumexp) per (bh, t).
// grid (8, NBH), block 128; wave owns a 32-row t tile (K frags hoisted),
// Q fragments shared across both 16-row halves.
__global__ void attn_stats_kernel(const _Float16* __restrict__ qT,
                                  const _Float16* __restrict__ kT,
                                  float* __restrict__ mstat,
                                  float* __restrict__ rstat) {
  int lane = threadIdx.x & 31, wv = threadIdx.x >> 5;
  int bh = blockIdx.y;
  int tb = (blockIdx.x * 4 + wv) * 32;
  const _Float16* qB = qT + (long)bh * NS * NCH;
  const _Float16* kB = kT + (long)bh * NS * NCH;
  v16h a00 = load_frag(kB + (long)tb * NCH, NCH, lane);
  v16h a01 = load_frag(kB + (long)tb * NCH + 32, NCH, lane);
  v16h a10 = load_frag(kB + (long)(tb + 16) * NCH, NCH, lane);
  v16h a11 = load_frag(kB + (long)(tb + 16) * NCH + 32, NCH, lane);
  float mrow[16], lrow[16];
#pragma unroll
  for (int e = 0; e < 16; ++e) { mrow[e] = -1e30f; lrow[e] = 0.f; }
  for (int sb = 0; sb < NS; sb += 16) {
    v16h b0 = load_frag(qB + (long)sb * NCH, NCH, lane);
    v16h b1 = load_frag(qB + (long)sb * NCH + 32, NCH, lane);
    v8f acc0 = {}, acc1 = {};
    acc0 = wmma16(a00, b0, acc0);
    acc0 = wmma16(a01, b1, acc0);
    acc1 = wmma16(a10, b0, acc1);
    acc1 = wmma16(a11, b1, acc1);
    // per-row (t) reduce over 16 s-columns; masks 1..8 stay in half-wave
#pragma unroll
    for (int half = 0; half < 2; ++half) {
      v8f acc = half ? acc1 : acc0;
#pragma unroll
      for (int e = 0; e < 8; ++e) {
        float v  = acc[e];
        float mx = v;
#pragma unroll
        for (int msk = 8; msk >= 1; msk >>= 1) mx = fmaxf(mx, __shfl_xor(mx, msk, 32));
        float ex = __expf(v - mx);
#pragma unroll
        for (int msk = 8; msk >= 1; msk >>= 1) ex += __shfl_xor(ex, msk, 32);
        int idx = half * 8 + e;
        float mnew = fmaxf(mrow[idx], mx);
        lrow[idx] = lrow[idx] * __expf(mrow[idx] - mnew) + ex * __expf(mx - mnew);
        mrow[idx] = mnew;
      }
    }
  }
  int hb8 = ((lane >> 4) & 1) * 8;
  if ((lane & 15) == 0) {
#pragma unroll
    for (int half = 0; half < 2; ++half)
#pragma unroll
      for (int e = 0; e < 8; ++e) {
        int t = tb + half * 16 + e + hb8;
        mstat[(long)bh * NS + t] = mrow[half * 8 + e];
        rstat[(long)bh * NS + t] = 1.f / lrow[half * 8 + e];
      }
  }
}

// ---------------- K5: out[c,s] = sum_t v[c,t] * P[t,s] -------------------
// Chained WMMA: score-WMMA f32 C layout == f16 B operand layout (rows -> K),
// so exp()+convert is lane-local; no LDS / shuffles in the hot loop.
// grid (8, NBH), block 128; wave owns a 32-wide s tile (K/V fragments and
// softmax stats shared across both 16-wide column halves), all 64 c rows.
__global__ void attn_pv_kernel(const _Float16* __restrict__ qT,
                               const _Float16* __restrict__ kT,
                               const _Float16* __restrict__ vbuf,
                               const float* __restrict__ mstat,
                               const float* __restrict__ rstat,
                               _Float16* __restrict__ oT) {
  int lane = threadIdx.x & 31, wv = threadIdx.x >> 5;
  int bh = blockIdx.y;
  int sb = (blockIdx.x * 4 + wv) * 32;
  const _Float16* qB = qT + (long)bh * NS * NCH;
  const _Float16* kB = kT + (long)bh * NS * NCH;
  const _Float16* vB = vbuf + (long)bh * NCH * NS;
  const float* mp = mstat + (long)bh * NS;
  const float* rp = rstat + (long)bh * NS;
  int hb8 = ((lane >> 4) & 1) * 8;
  // Q fragments are invariant over t: hoist (both column halves)
  v16h bq0a = load_frag(qB + (long)sb * NCH, NCH, lane);
  v16h bq1a = load_frag(qB + (long)sb * NCH + 32, NCH, lane);
  v16h bq0b = load_frag(qB + (long)(sb + 16) * NCH, NCH, lane);
  v16h bq1b = load_frag(qB + (long)(sb + 16) * NCH + 32, NCH, lane);
  v8f acca[4] = {}, accb[4] = {};
  for (int tb = 0; tb < NS; tb += 32) {
    __builtin_prefetch(kB + (long)(tb + 32) * NCH, 0, 1);
    v16h a00 = load_frag(kB + (long)tb * NCH, NCH, lane);
    v16h a01 = load_frag(kB + (long)tb * NCH + 32, NCH, lane);
    v16h a10 = load_frag(kB + (long)(tb + 16) * NCH, NCH, lane);
    v16h a11 = load_frag(kB + (long)(tb + 16) * NCH + 32, NCH, lane);
    v8f p0a = {}, p1a = {}, p0b = {}, p1b = {};
    p0a = wmma16(a00, bq0a, p0a);
    p0a = wmma16(a01, bq1a, p0a);
    p1a = wmma16(a10, bq0a, p1a);
    p1a = wmma16(a11, bq1a, p1a);
    p0b = wmma16(a00, bq0b, p0b);
    p0b = wmma16(a01, bq1b, p0b);
    p1b = wmma16(a10, bq0b, p1b);
    p1b = wmma16(a11, bq1b, p1b);
    // softmax stats for rows tb..tb+31: 32B vector loads, shared by both halves
    v8f m0 = *(const v8f*)(mp + tb + hb8);
    v8f r0 = *(const v8f*)(rp + tb + hb8);
    v8f m1 = *(const v8f*)(mp + tb + 16 + hb8);
    v8f r1 = *(const v8f*)(rp + tb + 16 + hb8);
    v16h pba, pbb;
#pragma unroll
    for (int e = 0; e < 8; ++e) {
      pba[e]     = (_Float16)(__expf(p0a[e] - m0[e]) * r0[e]);
      pba[e + 8] = (_Float16)(__expf(p1a[e] - m1[e]) * r1[e]);
      pbb[e]     = (_Float16)(__expf(p0b[e] - m0[e]) * r0[e]);
      pbb[e + 8] = (_Float16)(__expf(p1b[e] - m1[e]) * r1[e]);
    }
#pragma unroll
    for (int ct = 0; ct < 4; ++ct) {
      v16h av = load_frag(vB + (long)(ct * 16) * NS + tb, NS, lane);
      acca[ct] = wmma16(av, pba, acca[ct]);
      accb[ct] = wmma16(av, pbb, accb[ct]);
    }
  }
  // write oT[b][s][c] (c contiguous): v8h stores
  int b = bh / NHEADS, hd = bh % NHEADS;
  int n = lane & 15;
  _Float16* orow0 = oT + ((long)b * NS + sb + n) * NC + hd * NCH + hb8;
  _Float16* orow1 = oT + ((long)b * NS + sb + 16 + n) * NC + hd * NCH + hb8;
#pragma unroll
  for (int ct = 0; ct < 4; ++ct) {
    v8h pk0, pk1;
#pragma unroll
    for (int e = 0; e < 8; ++e) {
      pk0[e] = (_Float16)acca[ct][e];
      pk1[e] = (_Float16)accb[ct][e];
    }
    *(v8h*)(orow0 + ct * 16) = pk0;
    *(v8h*)(orow1 + ct * 16) = pk1;
  }
}

// ---------------- K6: y = w_out @ out + b_out + x (residual) -------------
// Roles swapped (M=s, N=o) so the fp32 epilogue is contiguous in s.
// Wave covers 32 s rows; weight (B) fragment shared across both halves.
// grid (8, 32, NB), block 128.
__global__ void attn_proj_kernel(const _Float16* __restrict__ wout16,
                                 const _Float16* __restrict__ oT,
                                 const float* __restrict__ bout,
                                 const float* __restrict__ x,
                                 float* __restrict__ y) {
  int lane = threadIdx.x & 31, wv = threadIdx.x >> 5;
  int sb = (blockIdx.x * 4 + wv) * 32;
  int ob = blockIdx.y * 16;
  int b  = blockIdx.z;
  const _Float16* oB = oT + (long)b * NS * NC;
  v8f acc0 = {}, acc1 = {};
  for (int kb = 0; kb < NC; kb += 32) {
    __builtin_prefetch(oB + (long)sb * NC + kb + 32, 0, 1);
    v16h a0 = load_frag(oB + (long)sb * NC + kb, NC, lane);        // A: oT[s][c]
    v16h a1 = load_frag(oB + (long)(sb + 16) * NC + kb, NC, lane);
    v16h bm = load_frag(wout16 + (long)ob * NC + kb, NC, lane);    // B: w[o][c]
    acc0 = wmma16(a0, bm, acc0);
    acc1 = wmma16(a1, bm, acc1);
  }
  int n = lane & 15, hb8 = ((lane >> 4) & 1) * 8;
  int o = ob + n;                      // lane owns one output channel
  float bo = bout[o];
  long base0 = ((long)b * NC + o) * NS + sb + hb8;       // 8 consecutive s
  long base1 = base0 + 16;
#pragma unroll
  for (int e = 0; e < 8; ++e) {
    y[base0 + e] = x[base0 + e] + acc0[e] + bo;
    y[base1 + e] = x[base1 + e] + acc1[e] + bo;
  }
}

// ---------------- host launcher ----------------
extern "C" void kernel_launch(void* const* d_in, const int* in_sizes, int n_in,
                              void* d_out, int out_size, void* d_ws, size_t ws_size,
                              hipStream_t stream) {
  (void)in_sizes; (void)n_in; (void)out_size; (void)ws_size;
  const float* x    = (const float*)d_in[0];
  const float* gsc  = (const float*)d_in[1];
  const float* gbi  = (const float*)d_in[2];
  const float* wqkv = (const float*)d_in[3];
  const float* bqkv = (const float*)d_in[4];
  const float* wout = (const float*)d_in[5];
  const float* bout = (const float*)d_in[6];
  float* out = (float*)d_out;

  size_t off = 0;
  auto wsalloc = [&](size_t bytes) -> void* {
    void* p = (void*)((char*)d_ws + off);
    off += (bytes + 255) & ~(size_t)255;
    return p;
  };
  _Float16* hT     = (_Float16*)wsalloc((size_t)NB * NS * NC * 2);        // 8 MB
  _Float16* wqkv16 = (_Float16*)wsalloc((size_t)3 * NC * NC * 2);         // 1.5 MB
  _Float16* wout16 = (_Float16*)wsalloc((size_t)NC * NC * 2);             // 0.5 MB
  _Float16* qT     = (_Float16*)wsalloc((size_t)NBH * NS * NCH * 2);      // 8 MB
  _Float16* kT     = (_Float16*)wsalloc((size_t)NBH * NS * NCH * 2);      // 8 MB
  _Float16* vbuf   = (_Float16*)wsalloc((size_t)NBH * NCH * NS * 2);      // 8 MB
  float*    mstat  = (float*)wsalloc((size_t)NBH * NS * 4);               // 256 KB
  float*    rstat  = (float*)wsalloc((size_t)NBH * NS * 4);               // 256 KB
  _Float16* oT     = (_Float16*)wsalloc((size_t)NB * NS * NC * 2);        // 8 MB

  attn_gn_kernel<<<dim3(NB * NGROUPS), dim3(256), 0, stream>>>(x, gsc, gbi, hT);
  attn_cvt_f16_kernel<<<dim3((3 * NC * NC + 255) / 256), dim3(256), 0, stream>>>(
      wqkv, wqkv16, 3 * NC * NC);
  attn_cvt_f16_kernel<<<dim3((NC * NC + 255) / 256), dim3(256), 0, stream>>>(
      wout, wout16, NC * NC);
  attn_qkv_gemm_kernel<<<dim3(8, 96, NB), dim3(128), 0, stream>>>(
      wqkv16, hT, bqkv, qT, kT, vbuf);
  attn_stats_kernel<<<dim3(8, NBH), dim3(128), 0, stream>>>(qT, kT, mstat, rstat);
  attn_pv_kernel<<<dim3(8, NBH), dim3(128), 0, stream>>>(
      qT, kT, vbuf, mstat, rstat, oT);
  attn_proj_kernel<<<dim3(8, 32, NB), dim3(128), 0, stream>>>(
      wout16, oT, bout, x, out);
}
